// Encoder_83494164234826
// MI455X (gfx1250) — matrix operand order repaired and checked by reading, not compiled, not run on hardware
//
#include <hip/hip_runtime.h>
#include <hip/hip_bf16.h>
#include <math.h>

typedef __attribute__((ext_vector_type(16))) __bf16 v16bf;
typedef __attribute__((ext_vector_type(8)))  float  v8f;

#define VOCAB 32000
#define EMB   256
#define UNITS 1024
#define NB    64
#define NTT   512
#define N3U   3072

union AFrag { v16bf v; unsigned int u[8]; };

__device__ __forceinline__ v8f wmma_bf16(v16bf a, v16bf b, v8f c) {
    return __builtin_amdgcn_wmma_f32_16x16x32_bf16(false, a, false, b, (short)0, c, false, false);
}

__device__ __forceinline__ float sigmoidf_(float x) {
    return 1.0f / (1.0f + __expf(-x));
}

// Gather a 16x32 bf16 A-fragment from a row-major [rows x ld] bf16 matrix.
// ISA 16-bit A layout: lane l -> row (l&15); dword d -> K pair at
// (d<4?0:16) + 8*(l>>4) + 2*(d&3).
__device__ __forceinline__ v16bf gather_a(const __bf16* __restrict__ base,
                                          size_t ld, int row0, int k0, int lane) {
    const __bf16* rp = base + (size_t)(row0 + (lane & 15)) * ld + k0;
    const unsigned int* p = (const unsigned int*)rp;
    const int hi = (lane >> 4) << 3;
    AFrag f;
#pragma unroll
    for (int d = 0; d < 8; ++d) {
        int koff = ((d & 4) << 2) + hi + ((d & 3) << 1);
        f.u[d] = p[koff >> 1];
    }
    return f.v;
}

// ---------------- prepack kernels ----------------

// Pack row-major f32 W[K x N] into WMMA B-operand tiles, tile order [nt][kt],
// per ISA B layout: lane l holds column nt*16+(l&15); its 16 contiguous bf16
// are K = kt*32 + 16*(s>>3) + 8*(l>>4) + (s&7)  (s = 0..15).
__global__ void pack_b_kernel(const float* __restrict__ W, int Ktiles, int N,
                              __bf16* __restrict__ out, int total) {
    int id = blockIdx.x * blockDim.x + threadIdx.x;
    if (id >= total) return;
    int s    = id & 15;
    int lane = (id >> 4) & 31;
    int tile = id >> 9;
    int nt = tile / Ktiles;
    int kt = tile - nt * Ktiles;
    int nn = (nt << 4) + (lane & 15);
    int kk = (kt << 5) + ((s & 8) << 1) + ((lane >> 4) << 3) + (s & 7);
    out[id] = (__bf16)W[(size_t)kk * N + nn];
}

__global__ void f32_to_bf16_kernel(const float* __restrict__ in,
                                   __bf16* __restrict__ out, int n) {
    int i = blockIdx.x * blockDim.x + threadIdx.x;
    if (i < n) out[i] = (__bf16)in[i];
}

__global__ void init_h_kernel(const float* __restrict__ hidden,
                              float* __restrict__ hf, __bf16* __restrict__ hbf, int n) {
    int i = blockIdx.x * blockDim.x + threadIdx.x;
    if (i < n) { float v = hidden[i]; hf[i] = v; hbf[i] = (__bf16)v; }
}

// ---------------- phase 1: xproj = emb_table[x] @ kernel + bias ----------------
// One wave computes a 16x64 block of xproj (4 C tiles share one A fragment).
__global__ __launch_bounds__(256) void xproj_gemm(const int* __restrict__ xk,
        const __bf16* __restrict__ embbf, const __bf16* __restrict__ kpack,
        const float* __restrict__ bias, float* __restrict__ xproj) {
    const int lane = threadIdx.x & 31;
    const int w  = (blockIdx.x << 3) + (threadIdx.x >> 5);   // 98304 waves
    const int mt = w / 48;                                   // 0..2047 row tiles
    const int ng = w - mt * 48;                              // 0..47 (64-col groups)
    const int hi = (lane >> 4) << 3;
    const int token = xk[(mt << 4) + (lane & 15)];
    const __bf16* arow = embbf + (size_t)token * EMB;

    v8f a0 = {}, a1 = {}, a2 = {}, a3 = {};
    for (int kt = 0; kt < 8; ++kt) {
        AFrag af;
        const unsigned int* p = (const unsigned int*)(arow + (kt << 5));
#pragma unroll
        for (int d = 0; d < 8; ++d) {
            int koff = ((d & 4) << 2) + hi + ((d & 3) << 1);
            af.u[d] = p[koff >> 1];
        }
        // n-tiles ng*4 + j, tile index = ntile*8 + kt, 512 elems/tile
        const __bf16* bp = kpack + (((size_t)ng * 32 + kt) << 9) + (lane << 4);
        v16bf b0 = *(const v16bf*)(bp);
        v16bf b1 = *(const v16bf*)(bp + 4096);
        v16bf b2 = *(const v16bf*)(bp + 8192);
        v16bf b3 = *(const v16bf*)(bp + 12288);
        a0 = wmma_bf16(af.v, b0, a0);
        a1 = wmma_bf16(af.v, b1, a1);
        a2 = wmma_bf16(af.v, b2, a2);
        a3 = wmma_bf16(af.v, b3, a3);
    }
    const int colbase = ng * 64 + (lane & 15);
    const float bi0 = bias[colbase], bi1 = bias[colbase + 16];
    const float bi2 = bias[colbase + 32], bi3 = bias[colbase + 48];
#pragma unroll
    for (int i = 0; i < 8; ++i) {
        size_t row = (size_t)(mt * 16 + hi + i);
        float* orow = xproj + row * N3U + colbase;
        orow[0]  = a0[i] + bi0;
        orow[16] = a1[i] + bi1;
        orow[32] = a2[i] + bi2;
        orow[48] = a3[i] + bi3;
    }
}

// ---------------- phase 2a: z/r gates for step t ----------------
// One wave: 16 batch rows x 16 unit cols; computes z-strip (rk col u) and
// r-strip (rk col 1024+u) with a shared h A-fragment; writes z and r*h.
__global__ __launch_bounds__(128) void gru_zr(const __bf16* __restrict__ hbf,
        const float* __restrict__ hf, const __bf16* __restrict__ rkp,
        const float* __restrict__ xproj, float* __restrict__ zbuf,
        __bf16* __restrict__ rhbf, int t) {
    const int lane = threadIdx.x & 31;
    const int w  = (blockIdx.x << 2) + (threadIdx.x >> 5);   // 256 waves
    const int mt = w >> 6;                                   // 0..3
    const int nt = w & 63;                                   // 0..63
    const int hi = (lane >> 4) << 3;
    const __bf16* bz = rkp + ((size_t)(nt)        * 32 << 9) + (lane << 4);
    const __bf16* br = rkp + ((size_t)(64 + nt)   * 32 << 9) + (lane << 4);
    v8f az = {}, ar = {};
    for (int kt = 0; kt < 32; ++kt) {
        v16bf af = gather_a(hbf, UNITS, mt << 4, kt << 5, lane);
        v16bf b0 = *(const v16bf*)(bz + ((size_t)kt << 9));
        v16bf b1 = *(const v16bf*)(br + ((size_t)kt << 9));
        az = wmma_bf16(af, b0, az);
        ar = wmma_bf16(af, b1, ar);
    }
    const int u = (nt << 4) + (lane & 15);
#pragma unroll
    for (int i = 0; i < 8; ++i) {
        int b = (mt << 4) + hi + i;
        const float* xr = xproj + (size_t)b * ((size_t)NTT * N3U) + (size_t)t * N3U;
        float z = sigmoidf_(xr[u] + az[i]);
        float r = sigmoidf_(xr[UNITS + u] + ar[i]);
        int idx = b * UNITS + u;
        zbuf[idx] = z;
        rhbf[idx] = (__bf16)(r * hf[idx]);
    }
}

// ---------------- phase 2b: candidate + state update for step t ----------------
__global__ __launch_bounds__(128) void gru_h(const __bf16* __restrict__ rhbf,
        const float* __restrict__ zbuf, const __bf16* __restrict__ rkp,
        const float* __restrict__ xproj, float* __restrict__ hf,
        __bf16* __restrict__ hbf, float* __restrict__ out,
        float* __restrict__ state, int t) {
    const int lane = threadIdx.x & 31;
    const int w  = (blockIdx.x << 2) + (threadIdx.x >> 5);   // 256 waves
    const int mt = w >> 6;
    const int nt = w & 63;
    const int hi = (lane >> 4) << 3;
    const __bf16* bh = rkp + ((size_t)(128 + nt) * 32 << 9) + (lane << 4);
    v8f ah = {};
    for (int kt = 0; kt < 32; ++kt) {
        v16bf af = gather_a(rhbf, UNITS, mt << 4, kt << 5, lane);
        v16bf b0 = *(const v16bf*)(bh + ((size_t)kt << 9));
        ah = wmma_bf16(af, b0, ah);
    }
    const int u = (nt << 4) + (lane & 15);
#pragma unroll
    for (int i = 0; i < 8; ++i) {
        int b = (mt << 4) + hi + i;
        const float* xr = xproj + (size_t)b * ((size_t)NTT * N3U) + (size_t)t * N3U;
        float hh = tanhf(xr[2 * UNITS + u] + ah[i]);
        int idx = b * UNITS + u;
        float z = zbuf[idx];
        float h = hf[idx];
        float hn = z * h + (1.0f - z) * hh;
        out[(size_t)b * ((size_t)NTT * UNITS) + (size_t)t * UNITS + u] = hn;
        hf[idx]  = hn;
        hbf[idx] = (__bf16)hn;
        if (t == NTT - 1) state[idx] = hn;
    }
}

extern "C" void kernel_launch(void* const* d_in, const int* in_sizes, int n_in,
                              void* d_out, int out_size, void* d_ws, size_t ws_size,
                              hipStream_t stream) {
    (void)in_sizes; (void)n_in; (void)out_size; (void)ws_size;
    const int*   x      = (const int*)d_in[0];
    const float* hidden = (const float*)d_in[1];
    const float* embT   = (const float*)d_in[2];
    const float* kern   = (const float*)d_in[3];
    const float* rk     = (const float*)d_in[4];
    const float* bias   = (const float*)d_in[5];
    float* out = (float*)d_out;

    char* ws = (char*)d_ws;
    __bf16* kpack  = (__bf16*)(ws);                  //  1.5 MB  (8*192 tiles)
    __bf16* rkpack = (__bf16*)(ws + 1572864);        //  6.0 MB  (32*192 tiles)
    __bf16* embbf  = (__bf16*)(ws + 7864320);        // 16.4 MB
    float*  xproj  = (float*) (ws + 24248320);       // 402.7 MB
    float*  zbuf   = (float*) (ws + 426901504);      // 256 KB
    __bf16* rhbf   = (__bf16*)(ws + 427163648);      // 128 KB
    float*  hf     = (float*) (ws + 427294720);      // 256 KB
    __bf16* hbf    = (__bf16*)(ws + 427556864);      // 128 KB

    {   // pack input-projection weights (K=256 -> 8 K-tiles)
        int total = 8 * 192 * 512;
        pack_b_kernel<<<(total + 255) / 256, 256, 0, stream>>>(kern, 8, N3U, kpack, total);
    }
    {   // pack recurrent weights (K=1024 -> 32 K-tiles)
        int total = 32 * 192 * 512;
        pack_b_kernel<<<(total + 255) / 256, 256, 0, stream>>>(rk, 32, N3U, rkpack, total);
    }
    {   int n = VOCAB * EMB;
        f32_to_bf16_kernel<<<(n + 255) / 256, 256, 0, stream>>>(embT, embbf, n);
    }
    {   int n = NB * UNITS;
        init_h_kernel<<<(n + 255) / 256, 256, 0, stream>>>(hidden, hf, hbf, n);
    }

    // phase 1: 32768x3072 xproj GEMM; 98304 waves, 8 waves per block
    xproj_gemm<<<12288, 256, 0, stream>>>(x, embbf, kpack, bias, xproj);

    // phase 2: 512 sequential steps, 2 launches/step (grid-wide sync via stream order)
    float* state = out + (size_t)NB * NTT * UNITS;
    for (int t = 0; t < NTT; ++t) {
        gru_zr<<<64, 128, 0, stream>>>(hbf, hf, rkpack, xproj, zbuf, rhbf, t);
        gru_h<<<64, 128, 0, stream>>>(rhbf, zbuf, rkpack, xproj, hf, hbf, out, state, t);
    }
}
